// AttentionFreeTransformer_15479062135477
// MI455X (gfx1250) — compile-verified, matrix operand order checked by reading, not run on hardware
//
#include <hip/hip_runtime.h>
#include <hip/hip_bf16.h>
#include <math.h>

#define BDIM 4
#define TDIM 4096
#define DDIM 2048
#define EDIM (3 * DDIM)      // 6144
#define BT   (BDIM * TDIM)   // 16384
#define EPS  1.1920929e-07f  // finfo(float32).eps
#define BK   32
#define CH   16              // scan chunks along T
#define CL   (TDIM / CH)     // 256

typedef __attribute__((ext_vector_type(16))) _Float16 v16h;
typedef __attribute__((ext_vector_type(8)))  float    v8f;
typedef __attribute__((ext_vector_type(4)))  int      v4i;

#if __has_builtin(__builtin_amdgcn_global_load_async_to_lds_b128)
#define USE_ASYNC_LDS 1
typedef __attribute__((address_space(1))) v4i gv4i;  // global src (non-const!)
typedef __attribute__((address_space(3))) v4i lv4i;  // LDS dst
#else
#define USE_ASYNC_LDS 0
#endif

static __device__ __forceinline__ void wait_async0() {
#if __has_builtin(__builtin_amdgcn_s_wait_asynccnt)
  __builtin_amdgcn_s_wait_asynccnt(0);
#else
  asm volatile("s_wait_asynccnt 0x0" ::: "memory");
#endif
}

// ---------- K1: rmsnorm(x) -> f16 xn ; passthrough copy x -> out ----------
__global__ __launch_bounds__(256)
void k_rmsnorm_x(const float* __restrict__ x, _Float16* __restrict__ xn,
                 float* __restrict__ out_x) {
  __shared__ float red[256];
  const int row = blockIdx.x;          // 0..BT-1
  const int t = threadIdx.x;
  const float4* xr = (const float4*)(x + (size_t)row * DDIM);
  float4 a = xr[t];
  float4 b = xr[t + 256];
  float s = a.x*a.x + a.y*a.y + a.z*a.z + a.w*a.w
          + b.x*b.x + b.y*b.y + b.z*b.z + b.w*b.w;
  red[t] = s; __syncthreads();
  #pragma unroll
  for (int o = 128; o > 0; o >>= 1) {
    if (t < o) red[t] += red[t + o];
    __syncthreads();
  }
  const float r = rsqrtf(red[0] * (1.0f / DDIM) + EPS);
  union { _Float16 h[4]; uint2 u; } ua, ub;
  ua.h[0] = (_Float16)(a.x * r); ua.h[1] = (_Float16)(a.y * r);
  ua.h[2] = (_Float16)(a.z * r); ua.h[3] = (_Float16)(a.w * r);
  ub.h[0] = (_Float16)(b.x * r); ub.h[1] = (_Float16)(b.y * r);
  ub.h[2] = (_Float16)(b.z * r); ub.h[3] = (_Float16)(b.w * r);
  uint2* xo = (uint2*)(xn + (size_t)row * DDIM);
  xo[t] = ua.u; xo[t + 256] = ub.u;
  float4* ox = (float4*)(out_x + (size_t)row * DDIM);
  ox[t] = a; ox[t + 256] = b;
}

// ---------- K2: cast weights f32 -> f16 ----------
__global__ __launch_bounds__(256)
void k_cast_w(const float* __restrict__ w, _Float16* __restrict__ wh) {
  const size_t i = (size_t)blockIdx.x * blockDim.x + threadIdx.x; // one float4
  const float4 v = ((const float4*)w)[i];
  union { _Float16 h[4]; uint2 u; } p;
  p.h[0] = (_Float16)v.x; p.h[1] = (_Float16)v.y;
  p.h[2] = (_Float16)v.z; p.h[3] = (_Float16)v.w;
  ((uint2*)wh)[i] = p.u;
}

// ---------- K3: WMMA GEMM  C[M=BT,N=EDIM] = A[M,K] * B[N,K]^T ----------
__global__ __launch_bounds__(256)
void k_gemm_wmma(const _Float16* __restrict__ A, const _Float16* __restrict__ Bm,
                 _Float16* __restrict__ C) {
  __shared__ _Float16 As[2][128 * BK];
  __shared__ _Float16 Bs[2][128 * BK];
  const int tid = threadIdx.x;
  const int lane = tid & 31, wid = tid >> 5;
  const int laneLo = lane & 15, laneHi = lane >> 4;
  const int wm = (wid & 1) * 64;    // 2 waves over M (64 rows each)
  const int wn = (wid >> 1) * 32;   // 4 waves over N (32 cols each)
  const int m0 = blockIdx.y * 128;
  const int n0 = blockIdx.x * 128;

  // per-thread copy assignment: 2 chunks of A-tile + 2 of B-tile, 16B each
  const int row0 = tid >> 2,          ch0 = tid & 3;
  const int row1 = (tid + 256) >> 2,  ch1 = tid & 3;

  v8f acc[4][2];
  #pragma unroll
  for (int i = 0; i < 4; ++i)
    #pragma unroll
    for (int j = 0; j < 2; ++j)
      #pragma unroll
      for (int e = 0; e < 8; ++e) acc[i][j][e] = 0.0f;

#if USE_ASYNC_LDS
  auto copy_tile = [&](int kt, int buf) {
    const int k0 = kt * BK;
    __builtin_amdgcn_global_load_async_to_lds_b128(
        (gv4i*)(A  + (size_t)(m0 + row0) * DDIM + k0 + ch0 * 8),
        (lv4i*)(&As[buf][row0 * BK + ch0 * 8]), 0, 0);
    __builtin_amdgcn_global_load_async_to_lds_b128(
        (gv4i*)(A  + (size_t)(m0 + row1) * DDIM + k0 + ch1 * 8),
        (lv4i*)(&As[buf][row1 * BK + ch1 * 8]), 0, 0);
    __builtin_amdgcn_global_load_async_to_lds_b128(
        (gv4i*)(Bm + (size_t)(n0 + row0) * DDIM + k0 + ch0 * 8),
        (lv4i*)(&Bs[buf][row0 * BK + ch0 * 8]), 0, 0);
    __builtin_amdgcn_global_load_async_to_lds_b128(
        (gv4i*)(Bm + (size_t)(n0 + row1) * DDIM + k0 + ch1 * 8),
        (lv4i*)(&Bs[buf][row1 * BK + ch1 * 8]), 0, 0);
  };
#else
  auto copy_tile = [&](int kt, int buf) {
    const int k0 = kt * BK;
    uint4 a0 = *(const uint4*)(A  + (size_t)(m0 + row0) * DDIM + k0 + ch0 * 8);
    uint4 a1 = *(const uint4*)(A  + (size_t)(m0 + row1) * DDIM + k0 + ch1 * 8);
    uint4 b0 = *(const uint4*)(Bm + (size_t)(n0 + row0) * DDIM + k0 + ch0 * 8);
    uint4 b1 = *(const uint4*)(Bm + (size_t)(n0 + row1) * DDIM + k0 + ch1 * 8);
    *(uint4*)(&As[buf][row0 * BK + ch0 * 8]) = a0;
    *(uint4*)(&As[buf][row1 * BK + ch1 * 8]) = a1;
    *(uint4*)(&Bs[buf][row0 * BK + ch0 * 8]) = b0;
    *(uint4*)(&Bs[buf][row1 * BK + ch1 * 8]) = b1;
  };
#endif

  auto compute = [&](int buf) {
    v16h bf0, bf1;
    {
      const int r0 = wn + laneLo, r1 = wn + 16 + laneLo;
      uint4* d0 = (uint4*)&bf0;
      d0[0] = *(const uint4*)(&Bs[buf][r0 * BK + laneHi * 8]);
      d0[1] = *(const uint4*)(&Bs[buf][r0 * BK + 16 + laneHi * 8]);
      uint4* d1 = (uint4*)&bf1;
      d1[0] = *(const uint4*)(&Bs[buf][r1 * BK + laneHi * 8]);
      d1[1] = *(const uint4*)(&Bs[buf][r1 * BK + 16 + laneHi * 8]);
    }
    #pragma unroll
    for (int mi = 0; mi < 4; ++mi) {
      v16h af;
      const int r = wm + mi * 16 + laneLo;
      uint4* d = (uint4*)&af;
      d[0] = *(const uint4*)(&As[buf][r * BK + laneHi * 8]);        // K 0..7 / 8..15
      d[1] = *(const uint4*)(&As[buf][r * BK + 16 + laneHi * 8]);   // K 16..23 / 24..31
      acc[mi][0] = __builtin_amdgcn_wmma_f32_16x16x32_f16(
          false, af, false, bf0, (short)0, acc[mi][0], false, false);
      acc[mi][1] = __builtin_amdgcn_wmma_f32_16x16x32_f16(
          false, af, false, bf1, (short)0, acc[mi][1], false, false);
    }
  };

  const int NK = DDIM / BK; // 64
  copy_tile(0, 0);
#if USE_ASYNC_LDS
  wait_async0();
#endif
  __syncthreads();

  for (int kt = 0; kt < NK; ++kt) {
    const int buf = kt & 1;
#if USE_ASYNC_LDS
    if (kt + 1 < NK) copy_tile(kt + 1, buf ^ 1);  // DMA overlaps with WMMAs below
    compute(buf);
    if (kt + 1 < NK) wait_async0();
    __syncthreads();
#else
    if (kt + 1 < NK) {                             // prefetch next tile into cache
      const int k0 = (kt + 1) * BK;
      __builtin_prefetch(A  + (size_t)(m0 + row0) * DDIM + k0 + ch0 * 8, 0, 0);
      __builtin_prefetch(Bm + (size_t)(n0 + row0) * DDIM + k0 + ch0 * 8, 0, 0);
    }
    compute(buf);
    if (kt + 1 < NK) copy_tile(kt + 1, buf ^ 1);
    __syncthreads();
#endif
  }

  // Epilogue: C/D layout -> lane (laneLo)=N col, VGPR j = M row (laneHi*8 + j)
  #pragma unroll
  for (int mi = 0; mi < 4; ++mi)
    #pragma unroll
    for (int ni = 0; ni < 2; ++ni) {
      const int col = n0 + wn + ni * 16 + laneLo;
      #pragma unroll
      for (int j = 0; j < 8; ++j) {
        const int row = m0 + wm + mi * 16 + laneHi * 8 + j;
        C[(size_t)row * EDIM + col] = (_Float16)acc[mi][ni][j];
      }
    }
}

// ---------- K4: per-row inverse RMS of q and k ----------
__global__ __launch_bounds__(256)
void k_qk_norm(const _Float16* __restrict__ qkv,
               float* __restrict__ rq, float* __restrict__ rk) {
  __shared__ float rs[256];
  const int row = blockIdx.x;
  const int t = threadIdx.x;
  const _Float16* base = qkv + (size_t)row * EDIM;
  float sq = 0.f, sk = 0.f;
  #pragma unroll
  for (int j = 0; j < 8; ++j) {
    const float qv = (float)base[t * 8 + j];
    const float kv = (float)base[DDIM + t * 8 + j];
    sq += qv * qv; sk += kv * kv;
  }
  rs[t] = sq; __syncthreads();
  #pragma unroll
  for (int o = 128; o > 0; o >>= 1) { if (t < o) rs[t] += rs[t + o]; __syncthreads(); }
  const float mq = rs[0]; __syncthreads();
  rs[t] = sk; __syncthreads();
  #pragma unroll
  for (int o = 128; o > 0; o >>= 1) { if (t < o) rs[t] += rs[t + o]; __syncthreads(); }
  const float mk = rs[0];
  if (t == 0) {
    rq[row] = rsqrtf(mq * (1.0f / DDIM) + EPS);
    rk[row] = rsqrtf(mk * (1.0f / DDIM) + EPS);
  }
}

// ---------- K5: per-chunk partial sums of w=exp(k') and w*v ----------
__global__ __launch_bounds__(256)
void k_scan_partial(const _Float16* __restrict__ qkv, const float* __restrict__ rk,
                    float* __restrict__ pw, float* __restrict__ pkv) {
  const int g = blockIdx.x * blockDim.x + threadIdx.x;  // B*CH*D
  const int d = g & (DDIM - 1);
  const int r = g >> 11;
  const int c = r & (CH - 1);
  const int b = r >> 4;
  float aw = 0.f, akv = 0.f;
  const int t0 = c * CL;
  for (int t = t0; t < t0 + CL; ++t) {
    const size_t bt = (size_t)(b * TDIM + t);
    const float kk = (float)qkv[bt * EDIM + DDIM + d] * rk[bt];
    const float vv = (float)qkv[bt * EDIM + 2 * DDIM + d];
    const float w = __expf(kk);
    aw += w; akv += w * vv;
  }
  pw [(size_t)r * DDIM + d] = aw;
  pkv[(size_t)r * DDIM + d] = akv;
}

// ---------- K6: exclusive prefix over chunks (in place) ----------
__global__ __launch_bounds__(256)
void k_scan_prefix(float* __restrict__ pw, float* __restrict__ pkv) {
  const int g = blockIdx.x * blockDim.x + threadIdx.x;  // B*D
  const int d = g & (DDIM - 1);
  const int b = g >> 11;
  float rw = 0.f, rkv = 0.f;
  for (int c = 0; c < CH; ++c) {
    const size_t i = ((size_t)(b * CH + c)) * DDIM + d;
    const float w = pw[i], kv = pkv[i];
    pw[i] = rw; pkv[i] = rkv;
    rw += w; rkv += kv;
  }
}

// ---------- K7: final scan + sigmoid(q)*y ----------
__global__ __launch_bounds__(256)
void k_scan_final(const _Float16* __restrict__ qkv,
                  const float* __restrict__ rq, const float* __restrict__ rk,
                  const float* __restrict__ pw, const float* __restrict__ pkv,
                  float* __restrict__ out) {
  const int g = blockIdx.x * blockDim.x + threadIdx.x;  // B*CH*D
  const int d = g & (DDIM - 1);
  const int r = g >> 11;
  const int c = r & (CH - 1);
  const int b = r >> 4;
  float aw  = pw [(size_t)r * DDIM + d];
  float akv = pkv[(size_t)r * DDIM + d];
  const int t0 = c * CL;
  for (int t = t0; t < t0 + CL; ++t) {
    const size_t bt = (size_t)(b * TDIM + t);
    const float kk = (float)qkv[bt * EDIM + DDIM + d] * rk[bt];
    const float vv = (float)qkv[bt * EDIM + 2 * DDIM + d];
    const float w = __expf(kk);
    aw += w; akv += w * vv;
    const float qq = (float)qkv[bt * EDIM + d] * rq[bt];
    const float y = akv / (aw + 1e-6f);
    const float sg = 1.0f / (1.0f + __expf(-qq));
    out[bt * DDIM + d] = sg * y;
  }
}

extern "C" void kernel_launch(void* const* d_in, const int* in_sizes, int n_in,
                              void* d_out, int out_size, void* d_ws, size_t ws_size,
                              hipStream_t stream) {
  const float* x = (const float*)d_in[0];
  const float* w = (const float*)d_in[1];
  float* out = (float*)d_out;   // [x passthrough | result], each BT*DDIM floats

  char* ws = (char*)d_ws;
  _Float16* xn  = (_Float16*)ws;                                        // BT*D f16
  _Float16* wh  = (_Float16*)(ws + (size_t)BT * DDIM * 2);              // E*D f16
  _Float16* qkv = (_Float16*)(ws + (size_t)BT * DDIM * 2
                                 + (size_t)EDIM * DDIM * 2);            // BT*E f16
  float* rq  = (float*)((char*)qkv + (size_t)BT * EDIM * 2);
  float* rk  = rq + BT;
  float* pw  = rk + BT;
  float* pkv = pw + (size_t)BDIM * CH * DDIM;

  k_rmsnorm_x<<<BT, 256, 0, stream>>>(x, xn, out);
  k_cast_w<<<(EDIM * DDIM / 4) / 256, 256, 0, stream>>>(w, wh);

  dim3 gg(EDIM / 128, BT / 128);   // 48 x 128 blocks
  k_gemm_wmma<<<gg, 256, 0, stream>>>(xn, wh, qkv);

  k_qk_norm<<<BT, 256, 0, stream>>>(qkv, rq, rk);
  k_scan_partial<<<(BDIM * CH * DDIM) / 256, 256, 0, stream>>>(qkv, rk, pw, pkv);
  k_scan_prefix<<<(BDIM * DDIM) / 256, 256, 0, stream>>>(pw, pkv);
  k_scan_final<<<(BDIM * CH * DDIM) / 256, 256, 0, stream>>>(
      qkv, rq, rk, pw, pkv, out + (size_t)BT * DDIM);
}
